// SphericalSweepStdMasked_23124103921904
// MI455X (gfx1250) — matrix operand order repaired and checked by reading, not compiled, not run on hardware
//
#include <hip/hip_runtime.h>
#include <hip/hip_bf16.h>

typedef __attribute__((ext_vector_type(16))) __bf16 v16bf;
typedef __attribute__((ext_vector_type(8)))  __bf16 v8bf;
typedef __attribute__((ext_vector_type(8)))  float  v8f;

// ---------------- geometry ----------------
constexpr int CAM = 6, C = 32, Hi = 128, Wi = 128;
constexpr int D = 16, H = 96, W = 96;            // X, Ho, Wo
constexpr int PD = D + 2, PH = H + 2, PW = W + 2; // zero-padded vol
constexpr int SPATIAL = D * H * W;               // 147456
constexpr int NTAPS = 27;
constexpr int PADVOL_ELEMS = PD * PH * PW * C;   // 5,531,904 bf16
constexpr int PACKW_ELEMS  = NTAPS * 2 * 32 * 16; // 27648 bf16 (tap, ntile, lane, elem)

// ---------------- workspace layout (bytes) ----------------
constexpr size_t OFF_PADVOL = 0;
constexpr size_t OFF_PACKW  = (size_t)PADVOL_ELEMS * 2;               // 11,063,808
constexpr size_t OFF_CONVO  = OFF_PACKW + (size_t)PACKW_ELEMS * 2;    // 11,119,104
constexpr size_t OFF_STATS  = OFF_CONVO + (size_t)C * SPATIAL * 4;    // 29,993,472
// total ws use: OFF_STATS + 512 bytes (~28.6 MB)

// ---------------- kernel 0: zero padded volume + global stats ----------------
__global__ void k_zero(uint32_t* __restrict__ padVolW, float* __restrict__ stats, int nWords) {
  int i = blockIdx.x * blockDim.x + threadIdx.x;
  for (int j = i; j < nWords; j += gridDim.x * blockDim.x) padVolW[j] = 0u;
  if (i < 64) stats[i] = 0.0f;
}

// ---------------- kernel 1: pack conv weights into ISA B-matrix lane layout ----
// packedW[((tap*2+nt)*32+lane)*16+e] = w[cout = nt*16+(lane&15)][cin = (e>>3)*16+(lane>=16?8:0)+(e&7)][tap]
__global__ void k_pack(const float* __restrict__ w, __bf16* __restrict__ packedW) {
  int i = blockIdx.x * 256 + threadIdx.x;
  if (i >= PACKW_ELEMS) return;
  int e    = i & 15;
  int lane = (i >> 4) & 31;
  int nt   = (i >> 9) & 1;
  int tap  = i >> 10;
  int cout = nt * 16 + (lane & 15);
  int cin  = ((e >> 3) * 16) + ((lane >> 4) * 8) + (e & 7);
  packedW[i] = (__bf16)w[(cout * C + cin) * NTAPS + tap];
}

// ---------------- kernel 2: grid-sample + masked moments across cameras -------
// one lane per (spatial point, channel); lanes of a wave = 32 channels of the
// same point, so mask math is wave-uniform and the channels-last store coalesces.
__global__ void k_sample(const float* __restrict__ feats, const float* __restrict__ grids,
                         const float* __restrict__ gmasks, const float* __restrict__ masks,
                         __bf16* __restrict__ padVol) {
  int tid = blockIdx.x * 256 + threadIdx.x;
  int c  = tid & 31;
  int sp = tid >> 5;                       // [0, SPATIAL)
  int wo = sp % W; int t = sp / W; int ho = t % H; int x = t / H;
  float sum = 0.f, sq = 0.f, cntf = 0.f;
  for (int cam = 0; cam < CAM; ++cam) {
    int gidx = ((cam * D + x) * H + ho) * W + wo;
    float g0 = grids[2 * gidx], g1 = grids[2 * gidx + 1];
    float gmv = gmasks[gidx];
    float gx = (g0 + 1.0f) * (0.5f * Wi) - 0.5f;
    float gy = (g1 + 1.0f) * (0.5f * Hi) - 0.5f;
    float x0f = floorf(gx), y0f = floorf(gy);
    float wx1 = gx - x0f, wy1 = gy - y0f;
    float wx0 = 1.0f - wx1, wy0 = 1.0f - wy1;
    int x0 = (int)x0f, y0 = (int)y0f, x1 = x0 + 1, y1 = y0 + 1;
    bool bx0 = (x0 >= 0) & (x0 < Wi), bx1 = (x1 >= 0) & (x1 < Wi);
    bool by0 = (y0 >= 0) & (y0 < Hi), by1 = (y1 >= 0) & (y1 < Hi);
    int xc0 = min(max(x0, 0), Wi - 1), xc1 = min(max(x1, 0), Wi - 1);
    int yc0 = min(max(y0, 0), Hi - 1), yc1 = min(max(y1, 0), Hi - 1);
    float w00 = wy0 * wx0 * (float)(by0 && bx0);
    float w01 = wy0 * wx1 * (float)(by0 && bx1);
    float w10 = wy1 * wx0 * (float)(by1 && bx0);
    float w11 = wy1 * wx1 * (float)(by1 && bx1);
    const float* mp = masks + cam * (Hi * Wi);
    float smv = w00 * mp[yc0 * Wi + xc0] + w01 * mp[yc0 * Wi + xc1]
              + w10 * mp[yc1 * Wi + xc0] + w11 * mp[yc1 * Wi + xc1];
    if ((smv > 0.0f) && (gmv > 0.0f)) {     // wave-uniform branch
      const float* fp = feats + ((size_t)cam * C + c) * (Hi * Wi);
      float v = w00 * fp[yc0 * Wi + xc0] + w01 * fp[yc0 * Wi + xc1]
              + w10 * fp[yc1 * Wi + xc0] + w11 * fp[yc1 * Wi + xc1];
      sum += v; sq += v * v; cntf += 1.0f;
    }
  }
  // moments form of the reference's masked mean / variance-with-fill
  bool valid = cntf > 1.0f;
  float cdiv = valid ? cntf : 1.0f;
  float avg  = sum / cdiv;
  float var  = valid ? fmaxf(sq / cdiv - avg * avg, 0.0f) : 0.0f;
  padVol[(((x + 1) * PH + (ho + 1)) * PW + (wo + 1)) * C + c] = (__bf16)var;
}

// ---------------- kernel 3: implicit-GEMM conv3d via WMMA bf16 ----------------
// one wave per 16-spatial x 32-channel output tile; K = 27 taps x 32 in-channels
__global__ void __launch_bounds__(256) k_conv(const __bf16* __restrict__ padVol,
                                              const __bf16* __restrict__ packedW,
                                              const float* __restrict__ conv_b,
                                              float* __restrict__ convOut,
                                              float* __restrict__ gstats) {
  __shared__ __attribute__((aligned(16))) __bf16 sW[PACKW_ELEMS]; // 55,296 B
  __shared__ float sStats[64];
  { // stage pre-swizzled weights to LDS
    const uint4* src = (const uint4*)packedW;
    uint4* dst = (uint4*)sW;
    for (int i = threadIdx.x; i < PACKW_ELEMS / 8; i += 256) dst[i] = src[i];
  }
  if (threadIdx.x < 64) sStats[threadIdx.x] = 0.0f;
  __syncthreads();

  int lane = threadIdx.x & 31;
  int wv   = threadIdx.x >> 5;
  int tile = blockIdx.x * 8 + wv;             // 9216 tiles, grid=1152 exact
  int w0 = (tile % 6) * 16; int tt = tile / 6; int h = tt % H; int d = tt / H;
  int m  = lane & 15;                         // A-matrix row (spatial w offset)
  int hi = lane >> 4;                         // lane half selects K sub-range

  v8f acc0 = {}; v8f acc1 = {};
  #pragma unroll
  for (int tap = 0; tap < NTAPS; ++tap) {
    int kd = tap / 9, kh = (tap / 3) % 3, kw = tap % 3;
    // padded coords: interior point p maps to p+1; sample offset k-1 => p+k
    int base = (((d + kd) * PH + (h + kh)) * PW + (w0 + m + kw)) * C;
    union { v16bf v; v8bf h2[2]; } A;
    // ISA 16-bit A 16x32: lanes<16 hold K 0..7 & 16..23, lanes>=16 hold 8..15 & 24..31
    A.h2[0] = *(const v8bf*)(padVol + base + hi * 8);
    A.h2[1] = *(const v8bf*)(padVol + base + 16 + hi * 8);
    union { v16bf v; v8bf h2[2]; } B0, B1;
    const v8bf* pb0 = (const v8bf*)(sW + ((tap * 2 + 0) * 32 + lane) * 16);
    const v8bf* pb1 = (const v8bf*)(sW + ((tap * 2 + 1) * 32 + lane) * 16);
    B0.h2[0] = pb0[0]; B0.h2[1] = pb0[1];
    B1.h2[0] = pb1[0]; B1.h2[1] = pb1[1];
    acc0 = __builtin_amdgcn_wmma_f32_16x16x32_bf16(false, A.v, false, B0.v,
                                                   (short)0, acc0, false, false);
    acc1 = __builtin_amdgcn_wmma_f32_16x16x32_bf16(false, A.v, false, B1.v,
                                                   (short)0, acc1, false, false);
  }

  // C/D layout: lanes<16: N=lane, M=r ; lanes>=16: N=lane-16, M=8+r
  int c0 = m, c1 = 16 + m;
  float b0 = conv_b[c0], b1 = conv_b[c1];
  int mb = hi * 8;
  float s0 = 0.f, q0 = 0.f, s1 = 0.f, q1 = 0.f;
  #pragma unroll
  for (int r = 0; r < 8; ++r) {
    int sp = d * (H * W) + h * W + w0 + mb + r;
    float v0 = acc0[r] + b0;
    float v1 = acc1[r] + b1;
    convOut[c0 * SPATIAL + sp] = v0;
    convOut[c1 * SPATIAL + sp] = v1;
    s0 += v0; q0 += v0 * v0; s1 += v1; q1 += v1 * v1;
  }
  atomicAdd(&sStats[c0], s0);
  atomicAdd(&sStats[32 + c0], q0);
  atomicAdd(&sStats[c1], s1);
  atomicAdd(&sStats[32 + c1], q1);
  __syncthreads();
  if (threadIdx.x < 64) atomicAdd(&gstats[threadIdx.x], sStats[threadIdx.x]);
}

// ---------------- kernel 4: finalize per-channel scale/shift ------------------
__global__ void k_finalize(const float* __restrict__ gamma, const float* __restrict__ beta,
                           float* __restrict__ stats) {
  int c = threadIdx.x;
  if (c < 32) {
    float n   = (float)SPATIAL;
    float mu  = stats[c] / n;
    float var = stats[32 + c] / n - mu * mu;
    float rs  = rsqrtf(var + 1e-5f);
    float sc  = gamma[c] * rs;
    stats[64 + c] = sc;
    stats[96 + c] = beta[c] - mu * sc;
  }
}

// ---------------- kernel 5: normalize + leaky ReLU ----------------------------
__global__ void k_norm(const float* __restrict__ convOut, const float* __restrict__ stats,
                       float* __restrict__ out) {
  int i = blockIdx.x * 256 + threadIdx.x;
  if (i >= C * SPATIAL) return;
  int c = i / SPATIAL;
  float v = convOut[i] * stats[64 + c] + stats[96 + c];
  out[i] = (v >= 0.0f) ? v : 0.01f * v;
}

extern "C" void kernel_launch(void* const* d_in, const int* in_sizes, int n_in,
                              void* d_out, int out_size, void* d_ws, size_t ws_size,
                              hipStream_t stream) {
  const float* feats  = (const float*)d_in[0];
  const float* grids  = (const float*)d_in[1];
  const float* gmasks = (const float*)d_in[2];
  const float* masks  = (const float*)d_in[3];
  const float* convw  = (const float*)d_in[4];
  const float* convb  = (const float*)d_in[5];
  const float* gamma  = (const float*)d_in[6];
  const float* beta   = (const float*)d_in[7];

  char* ws = (char*)d_ws;
  __bf16* padVol  = (__bf16*)(ws + OFF_PADVOL);
  __bf16* packedW = (__bf16*)(ws + OFF_PACKW);
  float*  convOut = (float*)(ws + OFF_CONVO);
  float*  stats   = (float*)(ws + OFF_STATS);
  float*  outp    = (float*)d_out;

  k_zero<<<2048, 256, 0, stream>>>((uint32_t*)padVol, stats, PADVOL_ELEMS / 2);
  k_pack<<<(PACKW_ELEMS + 255) / 256, 256, 0, stream>>>(convw, packedW);
  k_sample<<<(C * SPATIAL) / 256, 256, 0, stream>>>(feats, grids, gmasks, masks, padVol);
  k_conv<<<SPATIAL / 16 / 8, 256, 0, stream>>>(padVol, packedW, convb, convOut, stats);
  k_finalize<<<1, 64, 0, stream>>>(gamma, beta, stats);
  k_norm<<<(C * SPATIAL + 255) / 256, 256, 0, stream>>>(convOut, stats, outp);
}
